// Point_M2AE_SEG_37245956391390
// MI455X (gfx1250) — compile-verified, matrix operand order checked by reading, not run on hardware
//
#include <hip/hip_runtime.h>
#include <hip/hip_bf16.h>

typedef __attribute__((ext_vector_type(16))) __bf16 v16bf;
typedef __attribute__((ext_vector_type(8)))  float  v8f;

#define BATCH  8
#define NPTS   2048
#define NHEADS 6
#define CEILD(a,b) (((a)+(b)-1)/(b))

__device__ __forceinline__ float geluf(float x) {
    return 0.5f * x * (1.0f + erff(x * 0.70710678118654752440f));
}

// ---------------------------------------------------------------------------
// Furthest point sampling: one workgroup per batch, running min-dist array in
// LDS (fits easily in 320KB/WGP), tree argmax reduction per iteration.
// ---------------------------------------------------------------------------
__global__ void __launch_bounds__(256)
fps_kernel(const float* __restrict__ src, int M, int npoint, int* __restrict__ out)
{
    const int b = blockIdx.x;
    const float* p = src + (size_t)b * M * 3;
    __shared__ float dist[NPTS];
    __shared__ float rv[256];
    __shared__ int   ri[256];
    __shared__ int   curS;
    const int t = threadIdx.x;
    for (int i = t; i < M; i += 256) dist[i] = 1e10f;
    if (t == 0) curS = 0;
    __syncthreads();
    for (int it = 0; it < npoint; ++it) {
        const int cur = curS;
        if (t == 0) out[b * npoint + it] = cur;
        const float cx = p[cur*3+0], cy = p[cur*3+1], cz = p[cur*3+2];
        float bv = -1e30f; int bi = 0;
        for (int i = t; i < M; i += 256) {
            float dx = p[i*3+0]-cx, dy = p[i*3+1]-cy, dz = p[i*3+2]-cz;
            float dd = dx*dx + dy*dy + dz*dz;
            float dmin = fminf(dist[i], dd);
            dist[i] = dmin;
            if (dmin > bv) { bv = dmin; bi = i; }
        }
        rv[t] = bv; ri[t] = bi;
        __syncthreads();
        for (int sft = 128; sft > 0; sft >>= 1) {
            if (t < sft) {
                if (rv[t+sft] > rv[t] || (rv[t+sft] == rv[t] && ri[t+sft] < ri[t])) {
                    rv[t] = rv[t+sft]; ri[t] = ri[t+sft];
                }
            }
            __syncthreads();
        }
        if (t == 0) curS = ri[0];
        __syncthreads();
    }
}

__global__ void gather_centers(const float* __restrict__ src, const int* __restrict__ fidx,
                               float* __restrict__ ctr, int G, int M, int total)
{
    int i = blockIdx.x * blockDim.x + threadIdx.x;
    if (i >= total) return;
    int b = i / (G*3); int r = i - b*G*3; int g = r / 3; int c = r - g*3;
    ctr[i] = src[((size_t)b*M + fidx[b*G + g])*3 + c];
}

// thread per (b, center): insertion top-k (k<=16) over all source points
__global__ void knn_kernel(const float* __restrict__ ctr, const float* __restrict__ src,
                           int* __restrict__ out, int B, int G, int M, int K)
{
    int i = blockIdx.x * blockDim.x + threadIdx.x;
    if (i >= B*G) return;
    int b = i / G;
    const float* c = ctr + (size_t)i * 3;
    const float* p = src + (size_t)b * M * 3;
    const float cx = c[0], cy = c[1], cz = c[2];
    float bd[16]; int bi[16];
    for (int j = 0; j < K; ++j) { bd[j] = 1e30f; bi[j] = 0; }
    for (int m = 0; m < M; ++m) {
        float dx = p[m*3+0]-cx, dy = p[m*3+1]-cy, dz = p[m*3+2]-cz;
        float d = dx*dx + dy*dy + dz*dz;
        if (d < bd[K-1]) {
            int j = K - 1;
            while (j > 0 && bd[j-1] > d) { bd[j] = bd[j-1]; bi[j] = bi[j-1]; --j; }
            bd[j] = d; bi[j] = m;
        }
    }
    for (int j = 0; j < K; ++j) out[(size_t)i*K + j] = bi[j];
}

// stage-0 token input: gathered xyz relative to center (per batch)
__global__ void build_neigh(float* __restrict__ dst, const float* __restrict__ src,
                            const int* __restrict__ knn, const float* __restrict__ ctr,
                            int b, int G, int Kp, int M, int total)
{
    int i = blockIdx.x * blockDim.x + threadIdx.x;
    if (i >= total) return;
    int g = i / (Kp*3); int r = i - g*Kp*3; int k = r / 3; int c = r - k*3;
    int j = knn[((size_t)(b*G + g))*Kp + k];
    dst[i] = src[((size_t)b*M + j)*3 + c] - ctr[((size_t)(b*G + g))*3 + c];
}

// stage-1/2 token input: gathered previous-stage features (per batch)
__global__ void gather_feat(float* __restrict__ dst, const float* __restrict__ X,
                            const int* __restrict__ knn, int b, int G, int Kp, int C,
                            int Gprev, int total)
{
    int i = blockIdx.x * blockDim.x + threadIdx.x;
    if (i >= total) return;
    int g = i / (Kp*C); int r = i - g*Kp*C; int k = r / C; int c = r - k*C;
    int j = knn[((size_t)(b*G + g))*Kp + k];
    dst[i] = X[((size_t)b*Gprev + j)*C + c];
}

__global__ void maxpool_k(float* __restrict__ dst, const float* __restrict__ srcv,
                          int G, int Kp, int C, int total)
{
    int i = blockIdx.x * blockDim.x + threadIdx.x;
    if (i >= total) return;
    int g = i / C; int c = i - g*C;
    const float* s0 = srcv + (size_t)g*Kp*C + c;
    float m = s0[0];
    for (int k = 1; k < Kp; ++k) m = fmaxf(m, s0[(size_t)k*C]);
    dst[i] = m;
}

__global__ void concat_k(float* __restrict__ cat, const float* __restrict__ fg,
                         const float* __restrict__ f, int G, int Kp, int C2, int total)
{
    int i = blockIdx.x * blockDim.x + threadIdx.x;
    if (i >= total) return;
    int C2x2 = 2*C2;
    int g = i / (Kp*C2x2); int r = i - g*Kp*C2x2; int k = r / C2x2; int c = r - k*C2x2;
    cat[i] = (c < C2) ? fg[(size_t)g*C2 + c] : f[((size_t)g*Kp + k)*C2 + (c - C2)];
}

__global__ void add_kernel(float* __restrict__ y, const float* __restrict__ a,
                           const float* __restrict__ b, int n)
{
    int i = blockIdx.x * blockDim.x + threadIdx.x;
    if (i < n) y[i] = a[i] + b[i];
}

// wave32-native LayerNorm: one wave per row, shfl_xor reductions
__global__ void __launch_bounds__(256)
ln_kernel(const float* __restrict__ x, float* __restrict__ y,
          const float* __restrict__ gam, const float* __restrict__ bet, int rows, int d)
{
    const int wid = threadIdx.x >> 5, lane = threadIdx.x & 31;
    const int row = blockIdx.x * 8 + wid;
    if (row >= rows) return;
    const float* xr = x + (size_t)row * d;
    float s = 0.f, s2 = 0.f;
    for (int c = lane; c < d; c += 32) { float v = xr[c]; s += v; s2 += v*v; }
    for (int o = 16; o > 0; o >>= 1) { s += __shfl_xor(s, o, 32); s2 += __shfl_xor(s2, o, 32); }
    const float mean = s / d;
    const float var  = s2 / d - mean*mean;
    const float r = rsqrtf(var + 1e-5f);
    float* yr = y + (size_t)row * d;
    for (int c = lane; c < d; c += 32) yr[c] = (xr[c] - mean) * r * gam[c] + bet[c];
}

// masked softmax: one wave per (b,h,n) row; radius mask computed on the fly
// from centers (cdist >= radius -> -1e5), never materialized.
__global__ void __launch_bounds__(256)
attn_softmax(float* __restrict__ s, const float* __restrict__ ctr,
             float radius, int B, int H, int G)
{
    const int wid = threadIdx.x >> 5, lane = threadIdx.x & 31;
    const long long row = (long long)blockIdx.x * 8 + wid;
    const long long rows = (long long)B * H * G;
    if (row >= rows) return;
    const int n = (int)(row % G);
    const long long t = row / G;
    const int b = (int)(t / H);
    float* sr = s + row * G;
    const float* cb = ctr + (size_t)b * G * 3;
    const float cx = cb[n*3+0], cy = cb[n*3+1], cz = cb[n*3+2];
    float vals[16];                // G/32 <= 16
    float mx = -1e30f;
    int cnt = 0;
    for (int m = lane; m < G; m += 32) {
        float dx = cb[m*3+0]-cx, dy = cb[m*3+1]-cy, dz = cb[m*3+2]-cz;
        float d2 = dx*dx + dy*dy + dz*dz;
        float v = sr[m];
        if (sqrtf(fmaxf(d2, 0.f)) >= radius) v -= 100000.0f;
        vals[cnt++] = v; mx = fmaxf(mx, v);
    }
    for (int o = 16; o > 0; o >>= 1) mx = fmaxf(mx, __shfl_xor(mx, o, 32));
    float sum = 0.f; cnt = 0;
    for (int m = lane; m < G; m += 32) { float e = expf(vals[cnt] - mx); vals[cnt++] = e; sum += e; }
    for (int o = 16; o > 0; o >>= 1) sum += __shfl_xor(sum, o, 32);
    const float inv = 1.0f / sum;
    cnt = 0;
    for (int m = lane; m < G; m += 32) sr[m] = vals[cnt++] * inv;
}

// ---------------------------------------------------------------------------
// WMMA GEMM, wave per 16x(16*NT) tile, fp32 in/out, bf16 WMMA inside.
// NT=2 register-blocks two N tiles on one shared A fragment (2 WMMAs/K-step).
// BT=1: C = act(bn(alpha*A[M,K] @ W[N,K]^T + bias)) + resid   (W row-major)
// BT=0: B is [K,N] row-major (softmax @ V).
// K tails are handled branchlessly: the index is clamped in-bounds so loads
// stay unconditional (no EXEC manipulation -> EXEC is all-ones at the WMMA,
// and no per-element load/wait serialization), and the value is zero-selected.
// ---------------------------------------------------------------------------
template<int BT, int NT>
__global__ void __launch_bounds__(32)
gemm_wmma(const float* __restrict__ A, int lda, long long aSb, long long aSh,
          const float* __restrict__ Bm, int ldb, long long bSb, long long bSh,
          float* __restrict__ C, int ldc, long long cSb, long long cSh,
          const float* __restrict__ bias,
          const float* __restrict__ gamma, const float* __restrict__ beta,
          const float* __restrict__ resid, long long rSb, long long rSh,
          float alpha, int act, int M, int N, int K, int nH)
{
    const int lane = threadIdx.x & 31;
    const int hi   = lane >> 4;        // which K-interleave half this lane owns
    const int l16  = lane & 15;
    const int tn = blockIdx.x * (16 * NT);
    const int tm = blockIdx.y << 4;
    const int z  = blockIdx.z;
    const int bz = z / nH, hz = z - bz*nH;
    A  += (size_t)(bz*aSb + hz*aSh);
    Bm += (size_t)(bz*bSb + hz*bSh);
    C  += (size_t)(bz*cSb + hz*cSh);
    if (resid) resid += (size_t)(bz*rSb + hz*rSh);

    const bool fA = ((K & 31) == 0) && ((lda & 3) == 0);
    const bool fB = BT && ((K & 31) == 0) && ((ldb & 3) == 0);

    const int arow = tm + l16;
    const float* __restrict__ Ar = A + (size_t)arow * lda;

    v8f acc[NT];
    #pragma unroll
    for (int tt = 0; tt < NT; ++tt) acc[tt] = (v8f){0.f,0.f,0.f,0.f,0.f,0.f,0.f,0.f};

    for (int k0 = 0; k0 < K; k0 += 32) {
        const int ka = k0 + (hi << 3);     // frag elems 0..7
        const int kb = ka + 16;            // frag elems 8..15
        if (k0 + 32 < K)                   // global_prefetch_b8 for next A tile
            __builtin_prefetch(Ar + ka + 32, 0, 3);

        v16bf af;
        if (fA) {
            const float4* p = reinterpret_cast<const float4*>(Ar + ka);
            const float4* q = reinterpret_cast<const float4*>(Ar + kb);
            float4 x0 = p[0], x1 = p[1], y0 = q[0], y1 = q[1];
            af[0]=(__bf16)x0.x; af[1]=(__bf16)x0.y; af[2]=(__bf16)x0.z; af[3]=(__bf16)x0.w;
            af[4]=(__bf16)x1.x; af[5]=(__bf16)x1.y; af[6]=(__bf16)x1.z; af[7]=(__bf16)x1.w;
            af[8]=(__bf16)y0.x; af[9]=(__bf16)y0.y; af[10]=(__bf16)y0.z; af[11]=(__bf16)y0.w;
            af[12]=(__bf16)y1.x; af[13]=(__bf16)y1.y; af[14]=(__bf16)y1.z; af[15]=(__bf16)y1.w;
        } else {
            #pragma unroll
            for (int j = 0; j < 8; ++j) {
                int k = ka + j; int kc = (k < K) ? k : 0;
                float v = Ar[kc];                       // unconditional load
                af[j] = (__bf16)((k < K) ? v : 0.0f);   // v_cndmask select
            }
            #pragma unroll
            for (int j = 0; j < 8; ++j) {
                int k = kb + j; int kc = (k < K) ? k : 0;
                float v = Ar[kc];
                af[8+j] = (__bf16)((k < K) ? v : 0.0f);
            }
        }

        #pragma unroll
        for (int tt = 0; tt < NT; ++tt) {
            const int bcol = tn + tt*16 + l16;
            v16bf bfv;
            if (BT) {
                const float* __restrict__ Br = Bm + (size_t)bcol * ldb;
                if (fB) {
                    const float4* p = reinterpret_cast<const float4*>(Br + ka);
                    const float4* q = reinterpret_cast<const float4*>(Br + kb);
                    float4 x0 = p[0], x1 = p[1], y0 = q[0], y1 = q[1];
                    bfv[0]=(__bf16)x0.x; bfv[1]=(__bf16)x0.y; bfv[2]=(__bf16)x0.z; bfv[3]=(__bf16)x0.w;
                    bfv[4]=(__bf16)x1.x; bfv[5]=(__bf16)x1.y; bfv[6]=(__bf16)x1.z; bfv[7]=(__bf16)x1.w;
                    bfv[8]=(__bf16)y0.x; bfv[9]=(__bf16)y0.y; bfv[10]=(__bf16)y0.z; bfv[11]=(__bf16)y0.w;
                    bfv[12]=(__bf16)y1.x; bfv[13]=(__bf16)y1.y; bfv[14]=(__bf16)y1.z; bfv[15]=(__bf16)y1.w;
                } else {
                    #pragma unroll
                    for (int j = 0; j < 8; ++j) {
                        int k = ka + j; int kc = (k < K) ? k : 0;
                        float v = Br[kc];
                        bfv[j] = (__bf16)((k < K) ? v : 0.0f);
                    }
                    #pragma unroll
                    for (int j = 0; j < 8; ++j) {
                        int k = kb + j; int kc = (k < K) ? k : 0;
                        float v = Br[kc];
                        bfv[8+j] = (__bf16)((k < K) ? v : 0.0f);
                    }
                }
            } else {
                #pragma unroll
                for (int j = 0; j < 8; ++j) {
                    int k = ka + j; int kc = (k < K) ? k : 0;
                    float v = Bm[(size_t)kc*ldb + bcol];
                    bfv[j] = (__bf16)((k < K) ? v : 0.0f);
                }
                #pragma unroll
                for (int j = 0; j < 8; ++j) {
                    int k = kb + j; int kc = (k < K) ? k : 0;
                    float v = Bm[(size_t)kc*ldb + bcol];
                    bfv[8+j] = (__bf16)((k < K) ? v : 0.0f);
                }
            }
            acc[tt] = __builtin_amdgcn_wmma_f32_16x16x32_bf16(false, af, false, bfv,
                                                              (short)0, acc[tt], false, false);
        }
    }

    // Epilogue: alpha, bias, folded eval-BN (1/sqrt(1+eps)*gamma, beta), act, resid
    #pragma unroll
    for (int tt = 0; tt < NT; ++tt) {
        const int n = tn + tt*16 + l16;
        const float bv = bias ? bias[n] : 0.0f;
        float gv = 1.0f, bev = 0.0f;
        if (gamma) { gv = gamma[n]; bev = beta[n]; }
        #pragma unroll
        for (int j = 0; j < 8; ++j) {
            const int m = tm + (hi << 3) + j;
            float v = acc[tt][j] * alpha + bv;
            if (gamma) v = v * 0.99999500003749969f * gv + bev;
            if (act == 1)      v = fmaxf(v, 0.0f);
            else if (act == 2) v = geluf(v);
            if (resid) v += resid[(size_t)m*ldc + n];
            C[(size_t)m*ldc + n] = v;
        }
    }
}

static void gemm_launch(hipStream_t st, int BT,
                        const float* A, int lda, long long aSb, long long aSh,
                        const float* B, int ldb, long long bSb, long long bSh,
                        float* C, int ldc, long long cSb, long long cSh,
                        const float* bias, const float* gamma, const float* beta,
                        const float* resid, long long rSb, long long rSh,
                        float alpha, int act, int M, int N, int K, int nH, int nZ)
{
    const int NT = ((N & 31) == 0) ? 2 : 1;
    dim3 grid(N / (16*NT), M / 16, nZ), blk(32);
    if (BT) {
        if (NT == 2)
            gemm_wmma<1,2><<<grid, blk, 0, st>>>(A,lda,aSb,aSh, B,ldb,bSb,bSh, C,ldc,cSb,cSh,
                                                 bias,gamma,beta, resid,rSb,rSh, alpha,act,M,N,K,nH);
        else
            gemm_wmma<1,1><<<grid, blk, 0, st>>>(A,lda,aSb,aSh, B,ldb,bSb,bSh, C,ldc,cSb,cSh,
                                                 bias,gamma,beta, resid,rSb,rSh, alpha,act,M,N,K,nH);
    } else {
        if (NT == 2)
            gemm_wmma<0,2><<<grid, blk, 0, st>>>(A,lda,aSb,aSh, B,ldb,bSb,bSh, C,ldc,cSb,cSh,
                                                 bias,gamma,beta, resid,rSb,rSh, alpha,act,M,N,K,nH);
        else
            gemm_wmma<0,1><<<grid, blk, 0, st>>>(A,lda,aSb,aSh, B,ldb,bSb,bSh, C,ldc,cSb,cSh,
                                                 bias,gamma,beta, resid,rSb,rSh, alpha,act,M,N,K,nH);
    }
}

// ---------------------------------------------------------------------------
// Host orchestration
// ---------------------------------------------------------------------------
extern "C" void kernel_launch(void* const* d_in, const int* in_sizes, int n_in,
                              void* d_out, int out_size, void* d_ws, size_t ws_size,
                              hipStream_t stream)
{
    (void)in_sizes; (void)n_in; (void)out_size; (void)ws_size;
    const float* pts = (const float*)d_in[0];
    auto P = [&](int i) { return (const float*)d_in[i]; };

    // JAX pytree flattening (dicts alphabetical): pts, blocks(165), norms(6), pos(12), te(36)
    auto BLKI = [&](int li, int bi, int w) { return 1 + (li*5 + bi)*11 + w; };
    const int iF1=0,iF1B=1,iF2=2,iF2B=3,iLN1B=4,iLN1G=5,iLN2B=6,iLN2G=7,iPB=8,iPW=9,iQKV=10;
    auto NRM = [&](int li, int gb) { return 166 + li*2 + gb; };      // 0=b,1=g
    auto POS = [&](int li, int w)  { return 172 + li*4 + w;  };      // b1,b2,w1,w2
    auto TE  = [&](int li, int w)  { return 184 + li*12 + w; };      // b1,b2,b3,b4,be1,be3,g1,g3,w1,w2,w3,w4

    const int   DIMS_[3] = {96, 192, 384};
    const int   NG[3]    = {512, 256, 128};
    const int   GSZ[3]   = {16, 8, 8};
    const float RAD[3]   = {0.32f, 0.64f, 1.28f};

    size_t off = 0;
    auto allocF = [&](size_t n) { float* p = (float*)((char*)d_ws + off); off += ((n*sizeof(float)+255)/256)*256; return p; };
    auto allocI = [&](size_t n) { int*   p = (int*)  ((char*)d_ws + off); off += ((n*sizeof(int)  +255)/256)*256; return p; };

    float* ctr[3]; int* fidx[3]; int* knn[3];
    for (int s = 0; s < 3; ++s) {
        ctr[s]  = allocF((size_t)BATCH*NG[s]*3);
        fidx[s] = allocI((size_t)BATCH*NG[s]);
        knn[s]  = allocI((size_t)BATCH*NG[s]*GSZ[s]);
    }
    const size_t MD = (size_t)BATCH*512*96;           // = B*G*d, identical all stages
    float* Xbuf  = allocF(MD);
    float* posb  = allocF(MD);
    float* hbuf  = allocF(MD);
    float* h2b   = allocF(MD);
    float* lnb   = allocF(MD);
    float* attno = allocF(MD);
    float* qkvb  = allocF(MD*3);
    float* mlpb  = allocF(MD*4);
    float* scor  = allocF((size_t)BATCH*NHEADS*512*512);
    float* te_in = allocF(196608);
    float* te_y1 = allocF(1048576);
    float* te_y2 = allocF(2097152);
    float* te_fg = allocF(131072);
    float* te_ct = allocF(4194304);
    float* te_y3 = allocF(4194304);
    float* te_y4 = allocF(786432);

    // ---- hierarchical FPS + KNN grouping ----
    {
        const float* src = pts; int M = NPTS;
        for (int s = 0; s < 3; ++s) {
            fps_kernel<<<dim3(BATCH), dim3(256), 0, stream>>>(src, M, NG[s], fidx[s]);
            int tot = BATCH*NG[s]*3;
            gather_centers<<<CEILD(tot,256), 256, 0, stream>>>(src, fidx[s], ctr[s], NG[s], M, tot);
            int tc = BATCH*NG[s];
            knn_kernel<<<CEILD(tc,128), 128, 0, stream>>>(ctr[s], src, knn[s], BATCH, NG[s], M, GSZ[s]);
            src = ctr[s]; M = NG[s];
        }
    }

    size_t outOff = 0;
    for (int s = 0; s < 3; ++s) {
        const int d = DIMS_[s], G = NG[s], Kp = GSZ[s], Mrows = BATCH*G, hd = d/NHEADS;
        const int Cin = (s == 0) ? 3   : DIMS_[s-1];
        const int C1  = (s == 0) ? 128 : Cin;
        const int C2  = (s == 0) ? 256 : Cin;
        const int C3  = (s == 0) ? 512 : d;

        // ---- token embedding (per batch to keep workspace small; reads its
        //      own X region before overwriting it with the new tokens) ----
        for (int b = 0; b < BATCH; ++b) {
            const int Mt = G*Kp;
            if (s == 0) {
                int tot = G*Kp*3;
                build_neigh<<<CEILD(tot,256), 256, 0, stream>>>(te_in, pts, knn[0], ctr[0], b, G, Kp, NPTS, tot);
            } else {
                int tot = G*Kp*Cin;
                gather_feat<<<CEILD(tot,256), 256, 0, stream>>>(te_in, Xbuf, knn[s], b, G, Kp, Cin, NG[s-1], tot);
            }
            gemm_launch(stream,1, te_in,Cin,0,0, P(TE(s,8)),Cin,0,0, te_y1,C1,0,0,
                        P(TE(s,0)), P(TE(s,6)), P(TE(s,4)), nullptr,0,0, 1.0f,1, Mt,C1,Cin,1,1);
            gemm_launch(stream,1, te_y1,C1,0,0, P(TE(s,9)),C1,0,0, te_y2,C2,0,0,
                        P(TE(s,1)), nullptr,nullptr, nullptr,0,0, 1.0f,0, Mt,C2,C1,1,1);
            maxpool_k<<<CEILD(G*C2,256), 256, 0, stream>>>(te_fg, te_y2, G, Kp, C2, G*C2);
            concat_k<<<CEILD(G*Kp*2*C2,256), 256, 0, stream>>>(te_ct, te_fg, te_y2, G, Kp, C2, G*Kp*2*C2);
            gemm_launch(stream,1, te_ct,2*C2,0,0, P(TE(s,10)),2*C2,0,0, te_y3,C3,0,0,
                        P(TE(s,2)), P(TE(s,7)), P(TE(s,5)), nullptr,0,0, 1.0f,1, Mt,C3,2*C2,1,1);
            gemm_launch(stream,1, te_y3,C3,0,0, P(TE(s,11)),C3,0,0, te_y4,d,0,0,
                        P(TE(s,3)), nullptr,nullptr, nullptr,0,0, 1.0f,0, Mt,d,C3,1,1);
            maxpool_k<<<CEILD(G*d,256), 256, 0, stream>>>(Xbuf + (size_t)b*G*d, te_y4, G, Kp, d, G*d);
        }

        // ---- positional embedding: gelu(c@w1^T+b1)@w2^T+b2 ----
        gemm_launch(stream,1, ctr[s],3,0,0, P(POS(s,2)),3,0,0, lnb,d,0,0,
                    P(POS(s,0)), nullptr,nullptr, nullptr,0,0, 1.0f,2, Mrows,d,3,1,1);
        gemm_launch(stream,1, lnb,d,0,0, P(POS(s,3)),d,0,0, posb,d,0,0,
                    P(POS(s,1)), nullptr,nullptr, nullptr,0,0, 1.0f,0, Mrows,d,d,1,1);

        // ---- transformer blocks ----
        const float scal = 1.0f / sqrtf((float)hd);
        for (int bi = 0; bi < 5; ++bi) {
            add_kernel<<<CEILD(Mrows*d,256), 256, 0, stream>>>(hbuf, Xbuf, posb, Mrows*d);
            ln_kernel<<<CEILD(Mrows,8), 256, 0, stream>>>(hbuf, lnb, P(BLKI(s,bi,iLN1G)), P(BLKI(s,bi,iLN1B)), Mrows, d);
            gemm_launch(stream,1, lnb,d,0,0, P(BLKI(s,bi,iQKV)),d,0,0, qkvb,3*d,0,0,
                        nullptr,nullptr,nullptr, nullptr,0,0, 1.0f,0, Mrows,3*d,d,1,1);
            // scores = scale * Q @ K^T, batched over (b,h) directly on packed qkv
            gemm_launch(stream,1, qkvb,3*d,(long long)G*3*d,hd,
                        qkvb + d,3*d,(long long)G*3*d,hd,
                        scor,G,(long long)NHEADS*G*G,(long long)G*G,
                        nullptr,nullptr,nullptr, nullptr,0,0,
                        scal,0, G,G,hd, NHEADS, BATCH*NHEADS);
            attn_softmax<<<CEILD(BATCH*NHEADS*G,8), 256, 0, stream>>>(scor, ctr[s], RAD[s], BATCH, NHEADS, G);
            // O = softmax @ V, written straight into [B,G,H,hd] = [M,d] layout
            gemm_launch(stream,0, scor,G,(long long)NHEADS*G*G,(long long)G*G,
                        qkvb + 2*d,3*d,(long long)G*3*d,hd,
                        attno,d,(long long)G*d,hd,
                        nullptr,nullptr,nullptr, nullptr,0,0,
                        1.0f,0, G,hd,G, NHEADS, BATCH*NHEADS);
            gemm_launch(stream,1, attno,d,0,0, P(BLKI(s,bi,iPW)),d,0,0, h2b,d,0,0,
                        P(BLKI(s,bi,iPB)), nullptr,nullptr, hbuf,0,0, 1.0f,0, Mrows,d,d,1,1);
            ln_kernel<<<CEILD(Mrows,8), 256, 0, stream>>>(h2b, lnb, P(BLKI(s,bi,iLN2G)), P(BLKI(s,bi,iLN2B)), Mrows, d);
            gemm_launch(stream,1, lnb,d,0,0, P(BLKI(s,bi,iF1)),d,0,0, mlpb,4*d,0,0,
                        P(BLKI(s,bi,iF1B)), nullptr,nullptr, nullptr,0,0, 1.0f,2, Mrows,4*d,d,1,1);
            gemm_launch(stream,1, mlpb,4*d,0,0, P(BLKI(s,bi,iF2)),4*d,0,0, Xbuf,d,0,0,
                        P(BLKI(s,bi,iF2B)), nullptr,nullptr, h2b,0,0, 1.0f,0, Mrows,d,4*d,1,1);
        }
        // stage output: final LayerNorm straight into d_out
        ln_kernel<<<CEILD(Mrows,8), 256, 0, stream>>>(Xbuf, (float*)d_out + outOff,
                                                      P(NRM(s,1)), P(NRM(s,0)), Mrows, d);
        outOff += (size_t)Mrows * d;
    }
}
